// Quantizer_33672543601384
// MI455X (gfx1250) — matrix operand order repaired and checked by reading, not compiled, not run on hardware
//
#include <hip/hip_runtime.h>
#include <hip/hip_bf16.h>
#include <stddef.h>
#include <stdint.h>

// ---------------------------------------------------------------------------
// Problem constants (from the reference)
// ---------------------------------------------------------------------------
#define N_ROWS   8192
#define D_DIM    256
#define V_CODES  20000
#define K_DIM    4096
#define WORD_NUM 8
#define BLOCKS_B (N_ROWS / WORD_NUM)   // 1024
#define CTILES   (V_CODES / 16)        // 1250

typedef __attribute__((ext_vector_type(16))) __bf16 v16bf;
typedef __attribute__((ext_vector_type(8)))  float  v8f;
typedef __attribute__((ext_vector_type(4)))  unsigned int v4u;
typedef __attribute__((ext_vector_type(8)))  int    v8i;
typedef __attribute__((ext_vector_type(4)))  int    v4i;

__device__ __forceinline__ __bf16 f2bf(float f) { return (__bf16)f; }

// Load 8 contiguous floats -> bf16 into frag elements [e0 .. e0+7]
__device__ __forceinline__ void load8_cvt(const float* __restrict__ p, v16bf& frag, int e0) {
    const float4* q = reinterpret_cast<const float4*>(p);
    float4 f0 = q[0], f1 = q[1];
    frag[e0 + 0] = f2bf(f0.x); frag[e0 + 1] = f2bf(f0.y);
    frag[e0 + 2] = f2bf(f0.z); frag[e0 + 3] = f2bf(f0.w);
    frag[e0 + 4] = f2bf(f1.x); frag[e0 + 5] = f2bf(f1.y);
    frag[e0 + 6] = f2bf(f1.z); frag[e0 + 7] = f2bf(f1.w);
}

// ---------------------------------------------------------------------------
// Kernel 1: mapped = codebook_pca @ W_map^T + b_map   (V x D), f32 + bf16 copy
// One block per 16 V-rows covering the FULL D=256 strip: 8 waves x 2 column
// tiles. The 16 codebook rows are read once per block (WGP$ shares them
// across waves) -> codebook streams from HBM exactly once (~327 MB).
// ---------------------------------------------------------------------------
__global__ __launch_bounds__(256) void map_gemm_kernel(
    const float* __restrict__ cb,      // V x K
    const float* __restrict__ W,       // D x K
    const float* __restrict__ bmap,    // D
    float*  __restrict__ mapped,       // V x D
    __bf16* __restrict__ mapped_bf)    // V x D
{
    const int vbase = blockIdx.x * 16;          // 0..19984
    const int w     = threadIdx.x >> 5;         // wave in block, 0..7
    const int lane  = threadIdx.x & 31;
    const int m     = lane & 15;
    const int half  = lane >> 4;

    const float* arow  = cb + (size_t)(vbase + m) * K_DIM;
    const float* brow0 = W  + (size_t)(w * 32 + m) * K_DIM;        // cols w*32 .. +15
    const float* brow1 = W  + (size_t)(w * 32 + 16 + m) * K_DIM;   // cols w*32+16 .. +31

    v8f acc0 = {}, acc1 = {};
    for (int k0 = 0; k0 < K_DIM; k0 += 32) {
        v16bf a, b0, b1;
        load8_cvt(arow  + k0 + half * 16,      a,  0);
        load8_cvt(arow  + k0 + half * 16 + 8,  a,  8);
        load8_cvt(brow0 + k0 + half * 16,      b0, 0);
        load8_cvt(brow0 + k0 + half * 16 + 8,  b0, 8);
        load8_cvt(brow1 + k0 + half * 16,      b1, 0);
        load8_cvt(brow1 + k0 + half * 16 + 8,  b1, 8);
        acc0 = __builtin_amdgcn_wmma_f32_16x16x32_bf16(
            false, a, false, b0, (short)0, acc0, false, false);
        acc1 = __builtin_amdgcn_wmma_f32_16x16x32_bf16(
            false, a, false, b1, (short)0, acc1, false, false);
    }

    const int   col0  = w * 32 + m;
    const int   col1  = w * 32 + 16 + m;
    const float bias0 = bmap[col0];
    const float bias1 = bmap[col1];
#pragma unroll
    for (int r = 0; r < 8; ++r) {
        const int row = vbase + r + 8 * half;
        const float v0 = acc0[r] + bias0;
        const float v1 = acc1[r] + bias1;
        mapped   [(size_t)row * D_DIM + col0] = v0;
        mapped   [(size_t)row * D_DIM + col1] = v1;
        mapped_bf[(size_t)row * D_DIM + col0] = f2bf(v0);
        mapped_bf[(size_t)row * D_DIM + col1] = f2bf(v1);
    }
}

// ---------------------------------------------------------------------------
// Kernel 2: cb_sq[v] = sum_d mapped[v][d]^2   (one wave per row)
// ---------------------------------------------------------------------------
__global__ __launch_bounds__(256) void cbsq_kernel(
    const float* __restrict__ mapped, float* __restrict__ cb_sq)
{
    const int wave = (blockIdx.x * blockDim.x + threadIdx.x) >> 5;  // 0..19999
    const int lane = threadIdx.x & 31;
    const float* row = mapped + (size_t)wave * D_DIM;
    float s = 0.0f;
#pragma unroll
    for (int i = 0; i < D_DIM / 32; ++i) {
        const float v = row[lane + i * 32];
        s += v * v;
    }
#pragma unroll
    for (int off = 16; off > 0; off >>= 1) s += __shfl_down(s, off, 32);
    if (lane == 0) cb_sq[wave] = s;
}

// ---------------------------------------------------------------------------
// Kernel 3: score = cb_sq[col] - 2 * (z_e @ mapped^T); per-row top-8 (val,idx).
// One wave per 16-row strip of z_e (8 waves / block). All waves share the
// B tile (16 codes x 256 bf16 = 8KB), which wave 0 stages into a
// double-buffered LDS tile with the Tensor Data Mover (tensor_load_to_lds),
// synchronized with s_wait_tensorcnt + block barrier.
// ---------------------------------------------------------------------------

// Issue a TDM 2D tile load: mapped_bf[cbase .. cbase+15][0..255] -> LDS buf.
// D# built per CDNA5 ISA section 8.3/8.4.
__device__ __forceinline__ void tdm_load_tile(const __bf16* __restrict__ mapped_bf,
                                              int cbase, void* lds_dst)
{
    const uint64_t ga  = (uint64_t)(uintptr_t)(mapped_bf + (size_t)cbase * D_DIM);
    const uint32_t lds = (uint32_t)(uintptr_t)lds_dst;

    v4u g0;
    g0[0] = 1u;                                       // count=1, user descriptor
    g0[1] = lds;                                      // lds_addr
    g0[2] = (uint32_t)ga;                             // global_addr[31:0]
    g0[3] = (uint32_t)((ga >> 32) & 0x01FFFFFFu)      // global_addr[56:32]
          | (2u << 30);                               // type = 2 ("image")

    v8i g1;
    g1[0] = (1 << 16);                                // workgroup_mask=0, data_size=1 (2B)
    g1[1] = (D_DIM & 0xFFFF) << 16;                   // tensor_dim0[15:0]=256
    g1[2] = (0) | ((V_CODES & 0xFFFF) << 16);         // tensor_dim0 hi=0, tensor_dim1 lo=20000
    g1[3] = (V_CODES >> 16) | (D_DIM << 16);          // tensor_dim1 hi, tile_dim0=256
    g1[4] = 16;                                       // tile_dim1=16, tile_dim2=0
    g1[5] = D_DIM;                                    // tensor_dim0_stride[31:0]=256
    g1[6] = 0;                                        // stride hi / dim1_stride lo
    g1[7] = 0;

    v4i g2 = {0, 0, 0, 0};                            // 2D tensor: groups 2/3 unused
    v4i g3 = {0, 0, 0, 0};
    v8i g4 = {0, 0, 0, 0, 0, 0, 0, 0};                // extra group (clang-23 form)

    __builtin_amdgcn_tensor_load_to_lds(g0, g1, g2, g3, g4, 0);
}

__global__ __launch_bounds__(256) void dist_topk_kernel(
    const float*  __restrict__ z_e,
    const __bf16* __restrict__ mapped_bf,
    const float*  __restrict__ cb_sq,
    float* __restrict__ topv,     // N_ROWS x 8
    int*   __restrict__ topi)     // N_ROWS x 8
{
    __shared__ __align__(32) __bf16 btile[2][16][D_DIM];  // 2 x 8KB double buffer
    __shared__ float sc[8][16][17];                       // per-wave score tile (+pad)

    const int wlocal = threadIdx.x >> 5;
    const int wave = (blockIdx.x * blockDim.x + threadIdx.x) >> 5;  // 0..511
    const int lane = threadIdx.x & 31;
    const int m = lane & 15;
    const int half = lane >> 4;
    const int rbase = wave * 16;

    // Preload A fragments for all 8 K-steps (z_e rows, f32 -> bf16); they
    // stay resident in VGPRs for the whole 1250-tile column sweep.
    v16bf afrag[8];
    const float* arow = z_e + (size_t)(rbase + m) * D_DIM;
#pragma unroll
    for (int t = 0; t < 8; ++t) {
        const int k0 = t * 32;
        load8_cvt(arow + k0 + half * 16,     afrag[t], 0);
        load8_cvt(arow + k0 + half * 16 + 8, afrag[t], 8);
    }

    float tv[8];
    int   ti[8];
#pragma unroll
    for (int j = 0; j < 8; ++j) { tv[j] = 3.0e38f; ti[j] = 0; }

    // Prologue: stage tile 0 into buffer 0.
    if (wlocal == 0) tdm_load_tile(mapped_bf, 0, &btile[0][0][0]);

    for (int ct = 0; ct < CTILES; ++ct) {
        const int cbase = ct * 16;
        const int buf = ct & 1;

        if (wlocal == 0) __builtin_amdgcn_s_wait_tensorcnt(0);
        __syncthreads();   // tile ct visible to all waves; old reads of buf^1 done

        // Stage tile ct+1 into the other buffer while computing tile ct.
        if (wlocal == 0 && (ct + 1) < CTILES)
            tdm_load_tile(mapped_bf, cbase + 16, &btile[buf ^ 1][0][0]);

        const __bf16* bb = &btile[buf][m][16 * half];
        v8f acc = {};
#pragma unroll
        for (int t = 0; t < 8; ++t) {
            const v16bf b = *reinterpret_cast<const v16bf*>(bb + 32 * t);
            acc = __builtin_amdgcn_wmma_f32_16x16x32_bf16(
                false, afrag[t], false, b, (short)0, acc, false, false);
        }

        const float cs = cb_sq[cbase + m];   // column m of this tile
#pragma unroll
        for (int r = 0; r < 8; ++r) {
            sc[wlocal][r + 8 * half][m] = cs - 2.0f * acc[r];
        }
        __syncthreads();

        if (lane < 16) {
#pragma unroll
            for (int c = 0; c < 16; ++c) {
                float cv = sc[wlocal][lane][c];
                int   ci = cbase + c;
                if (cv < tv[7]) {
#pragma unroll
                    for (int j = 0; j < 8; ++j) {
                        if (cv < tv[j]) {
                            float ftmp = tv[j]; tv[j] = cv; cv = ftmp;
                            int   itmp = ti[j]; ti[j] = ci; ci = itmp;
                        }
                    }
                }
            }
        }
        __syncthreads();
    }

    if (lane < 16) {
        const int row = rbase + lane;
#pragma unroll
        for (int j = 0; j < 8; ++j) {
            topv[(size_t)row * 8 + j] = tv[j];
            topi[(size_t)row * 8 + j] = ti[j];
        }
    }
}

// ---------------------------------------------------------------------------
// Kernel 4: greedy per-block selection (8 words, exclusion), gather z_q,
// write outputs, per-block loss partial. One block per word-block.
// ---------------------------------------------------------------------------
__global__ __launch_bounds__(256) void select_gather_kernel(
    const float* __restrict__ z_e,
    const float* __restrict__ mapped,
    const float* __restrict__ topv,
    const int*   __restrict__ topi,
    const int*   __restrict__ token_id,
    float* __restrict__ out_zq,     // N x D
    int*   __restrict__ out_tok,    // N
    float* __restrict__ out_ztq,    // N x D (copy of z_e / residual)
    float* __restrict__ partials)   // BLOCKS_B
{
    __shared__ int   flat[WORD_NUM];
    __shared__ float red[256];

    const int b = blockIdx.x;        // 0..1023
    const int tid = threadIdx.x;

    if (tid == 0) {
        int chosen[WORD_NUM];
        for (int j = 0; j < WORD_NUM; ++j) {
            const int row = b * WORD_NUM + j;
            int pick = -1;
            for (int c = 0; c < 8 && pick < 0; ++c) {
                const int cand = topi[(size_t)row * 8 + c];
                bool used = false;
                for (int q = 0; q < j; ++q) used = used || (chosen[q] == cand);
                if (!used) pick = cand;
            }
            if (pick < 0) pick = topi[(size_t)row * 8];   // unreachable guard
            chosen[j] = pick;
            flat[j] = pick;
        }
    }
    __syncthreads();

    float lsum = 0.0f;
#pragma unroll
    for (int e = tid; e < WORD_NUM * D_DIM; e += 256) {
        const int r = e >> 8;             // word in block
        const int d = e & (D_DIM - 1);
        const int row = b * WORD_NUM + r;
        const float zq = mapped[(size_t)flat[r] * D_DIM + d];
        const float ze = z_e[(size_t)row * D_DIM + d];
        out_zq [(size_t)row * D_DIM + d] = zq;   // z_q_st == z_q numerically
        out_ztq[(size_t)row * D_DIM + d] = ze;   // residual at layer 0 == z_e
        const float df = zq - ze;
        lsum += df * df;
    }
    if (tid < WORD_NUM) out_tok[b * WORD_NUM + tid] = token_id[flat[tid]];

    red[tid] = lsum;
    __syncthreads();
    for (int s = 128; s > 0; s >>= 1) {
        if (tid < s) red[tid] += red[tid + s];
        __syncthreads();
    }
    if (tid == 0) partials[b] = red[0];
}

// ---------------------------------------------------------------------------
// Kernel 5: deterministic loss reduction (fixed order).
// loss = 0.75*vq + 0.25*commit = mean((z_q - z_e)^2)
// ---------------------------------------------------------------------------
__global__ __launch_bounds__(256) void loss_reduce_kernel(
    const float* __restrict__ partials, float* __restrict__ loss_out)
{
    __shared__ float red[256];
    const int tid = threadIdx.x;
    float s = 0.0f;
    for (int i = tid; i < BLOCKS_B; i += 256) s += partials[i];
    red[tid] = s;
    __syncthreads();
    for (int st = 128; st > 0; st >>= 1) {
        if (tid < st) red[tid] += red[tid + st];
        __syncthreads();
    }
    if (tid == 0) *loss_out = red[0] / (float)((size_t)N_ROWS * D_DIM);
}

// ---------------------------------------------------------------------------
// Host-side launcher
// ---------------------------------------------------------------------------
extern "C" void kernel_launch(void* const* d_in, const int* in_sizes, int n_in,
                              void* d_out, int out_size, void* d_ws, size_t ws_size,
                              hipStream_t stream)
{
    (void)in_sizes; (void)n_in; (void)out_size; (void)ws_size;

    const float* z_e   = (const float*)d_in[0];   // N x D
    const float* cb    = (const float*)d_in[1];   // V x K
    const float* W     = (const float*)d_in[2];   // D x K
    const float* bmap  = (const float*)d_in[3];   // D
    const int*   tok   = (const int*)  d_in[4];   // V

    // Workspace layout (bytes)
    char* ws = (char*)d_ws;
    size_t off = 0;
    float*  mapped    = (float*) (ws + off); off += (size_t)V_CODES * D_DIM * 4;   // 20.48 MB
    __bf16* mapped_bf = (__bf16*)(ws + off); off += (size_t)V_CODES * D_DIM * 2;   // 10.24 MB
    float*  cb_sq     = (float*) (ws + off); off += (size_t)V_CODES * 4;
    float*  topv      = (float*) (ws + off); off += (size_t)N_ROWS * 8 * 4;
    int*    topi      = (int*)   (ws + off); off += (size_t)N_ROWS * 8 * 4;
    float*  partials  = (float*) (ws + off); off += (size_t)BLOCKS_B * 4;

    // Output layout: [z_q_st (N*D)] [loss (1)] [token gather (N, int bits)] [z_e copy (N*D)]
    float* out      = (float*)d_out;
    float* out_zq   = out;
    float* loss_out = out + (size_t)N_ROWS * D_DIM;
    int*   out_tok  = (int*)(out + (size_t)N_ROWS * D_DIM + 1);
    float* out_ztq  = out + (size_t)N_ROWS * D_DIM + 1 + N_ROWS;

    // 1. mapped = codebook @ W^T + b  (one block per 16 codebook rows)
    map_gemm_kernel<<<CTILES, 256, 0, stream>>>(cb, W, bmap, mapped, mapped_bf);
    // 2. cb_sq  (20000 rows, 1 wave each)
    cbsq_kernel<<<2500, 256, 0, stream>>>(mapped, cb_sq);
    // 3. scores + per-row top-8 (512 waves; TDM-staged B tiles)
    dist_topk_kernel<<<64, 256, 0, stream>>>(z_e, mapped_bf, cb_sq, topv, topi);
    // 4. greedy selection + gather + partial loss
    select_gather_kernel<<<BLOCKS_B, 256, 0, stream>>>(
        z_e, mapped, topv, topi, tok, out_zq, out_tok, out_ztq, partials);
    // 5. deterministic loss reduction
    loss_reduce_kernel<<<1, 256, 0, stream>>>(partials, loss_out);
}
